// CircuitGnn_45037027066128
// MI455X (gfx1250) — matrix operand order repaired and checked by reading, not compiled, not run on hardware
//
#include <hip/hip_runtime.h>
#include <hip/hip_bf16.h>
#include <float.h>

typedef __attribute__((ext_vector_type(16))) __bf16 v16bf;
typedef __attribute__((ext_vector_type(8)))  float  v8f;

#define FEAT 64
#define SLOPE 0.01f
#define GN_EPS 1e-5f
#define TILES_PER_WAVE 4

// ---------------- degree ----------------
__global__ void deg_kernel(const int* __restrict__ dst, float* __restrict__ deg, int n_edges) {
    int e = blockIdx.x * blockDim.x + threadIdx.x;
    if (e >= n_edges) return;
    atomicAdd(&deg[dst[e]], 1.0f);
}

__global__ void invdeg_kernel(float* __restrict__ deg, int n_nodes) {
    int i = blockIdx.x * blockDim.x + threadIdx.x;
    if (i >= n_nodes) return;
    deg[i] = 1.0f / fmaxf(deg[i], 1.0f);
}

// ---------------- edge scatter (mean-agg numerator) ----------------
// 16 threads per edge, 4 channels each (float4 gather + 4 f32 atomics)
__global__ void scatter_kernel(const float* __restrict__ h, const int* __restrict__ src,
                               const int* __restrict__ dst, float* __restrict__ agg, int n_edges) {
    long long gid = (long long)blockIdx.x * blockDim.x + threadIdx.x;
    int e  = (int)(gid >> 4);
    int c4 = ((int)gid & 15) * 4;
    if (e >= n_edges) return;
    int s = src[e], d = dst[e];
    const float4 v = *(const float4*)(h + (long long)s * FEAT + c4);
    float* base = agg + (long long)d * FEAT + c4;
    atomicAdd(base + 0, v.x);
    atomicAdd(base + 1, v.y);
    atomicAdd(base + 2, v.z);
    atomicAdd(base + 3, v.w);
}

// ---------------- SAGE layer GEMM: out = (agg*invdeg)@Wl^T + b + h@Wr^T, leaky+residual ----------------
// 8 waves/block; each wave owns TILES_PER_WAVE consecutive 16-row tiles (amortizes LDS weight fill).
// Per tile: N=64 -> 4 col tiles, 4 chained bf16 WMMAs each.
__global__ __launch_bounds__(256) void sage_gemm_kernel(
    const float* __restrict__ hin, const float* __restrict__ agg,
    const float* __restrict__ invdeg,
    const float* __restrict__ Wl, const float* __restrict__ bl, const float* __restrict__ Wr,
    float* __restrict__ hout, int n_nodes)
{
    __shared__ __align__(32) __bf16 wl_s[FEAT * FEAT];
    __shared__ __align__(32) __bf16 wr_s[FEAT * FEAT];
    __shared__ float b_s[FEAT];

    int tid = threadIdx.x;
    for (int i = tid; i < FEAT * FEAT; i += 256) {
        wl_s[i] = (__bf16)Wl[i];        // stored [n][k] row-major => B[k][n] fragment reads contiguous in k
        wr_s[i] = (__bf16)Wr[i];
    }
    if (tid < FEAT) b_s[tid] = bl[tid];
    __syncthreads();

    int wave = tid >> 5;
    int lane = tid & 31;
    int half = lane >> 4;                       // K-subrange selector per ISA A/B layouts
    int lan16 = lane & 15;
    int tile0 = (blockIdx.x * 8 + wave) * TILES_PER_WAVE;

    for (int tt = 0; tt < TILES_PER_WAVE; ++tt) {
        int tile = tile0 + tt;
        if (tile * 16 >= n_nodes) break;        // wave-uniform (EXEC stays all-ones for WMMA)
        bool full = (tile * 16 + 16) <= n_nodes; // wave-uniform: unguarded epilogue on common path

        int row   = tile * 16 + lan16;
        int row_c = min(row, n_nodes - 1);      // clamp for partial tiles (loads only)
        float idg = invdeg[row_c];
        const float* hrow = hin + (long long)row_c * FEAT;
        const float* arow = agg + (long long)row_c * FEAT;

        // A fragments: contiguous k order (same as B). lane half h: element i -> K = kb*32 + h*16 + i
        v16bf aH[2], aA[2];
#pragma unroll
        for (int kb = 0; kb < 2; ++kb) {
            int k0 = kb * 32 + half * 16;
#pragma unroll
            for (int j = 0; j < 8; ++j) {
                aH[kb][j]     = (__bf16)hrow[k0 + j];
                aH[kb][j + 8] = (__bf16)hrow[k0 + 8 + j];
                aA[kb][j]     = (__bf16)(arow[k0 + j] * idg);
                aA[kb][j + 8] = (__bf16)(arow[k0 + 8 + j] * idg);
            }
        }

#pragma unroll
        for (int ct = 0; ct < 4; ++ct) {
            int n = ct * 16 + lan16;            // this lane's output column
            v8f acc = {};
#pragma unroll
            for (int kb = 0; kb < 2; ++kb) {
                // B fragment: contiguous k order (same as A): element i -> K = kb*32 + half*16 + i (32B LDS read)
                int koff = kb * 32 + half * 16;
                v16bf bwl = *(const v16bf*)&wl_s[n * FEAT + koff];
                v16bf bwr = *(const v16bf*)&wr_s[n * FEAT + koff];
                acc = __builtin_amdgcn_wmma_f32_16x16x32_bf16(false, aA[kb], false, bwl, (short)0, acc, false, false);
                acc = __builtin_amdgcn_wmma_f32_16x16x32_bf16(false, aH[kb], false, bwr, (short)0, acc, false, false);
            }
            float bias = b_s[n];
            float* outb = hout + (long long)(tile * 16 + half * 8) * FEAT + n;
            if (full) {
#pragma unroll
                for (int j = 0; j < 8; ++j) {   // C/D layout: VGPR j, lane half -> row
                    float v  = acc[j] + bias;
                    float lr = v > 0.0f ? v : SLOPE * v;
                    outb[(long long)j * FEAT] = lr + v;
                }
            } else {
#pragma unroll
                for (int j = 0; j < 8; ++j) {
                    int m = tile * 16 + half * 8 + j;
                    if (m < n_nodes) {
                        float v  = acc[j] + bias;
                        float lr = v > 0.0f ? v : SLOPE * v;
                        hout[(long long)m * FEAT + n] = lr + v;
                    }
                }
            }
        }
    }
}

// ---------------- GraphNorm ----------------
__global__ void gn_stats_kernel(const float* __restrict__ x, const int* __restrict__ batch,
                                float* __restrict__ gsum, float* __restrict__ gcnt, int n_nodes) {
    long long gid = (long long)blockIdx.x * blockDim.x + threadIdx.x;
    int node = (int)(gid >> 4);
    int c4   = ((int)gid & 15) * 4;
    if (node >= n_nodes) return;
    int g = batch[node];
    float4 v = *(const float4*)(x + (long long)node * FEAT + c4);
    float* s = gsum + (long long)g * FEAT + c4;
    atomicAdd(s + 0, v.x); atomicAdd(s + 1, v.y); atomicAdd(s + 2, v.z); atomicAdd(s + 3, v.w);
    if (((int)gid & 15) == 0) atomicAdd(&gcnt[g], 1.0f);
}

__global__ void gn_mean_kernel(float* __restrict__ gsum, const float* __restrict__ gcnt, int total) {
    int i = blockIdx.x * blockDim.x + threadIdx.x;
    if (i >= total) return;
    gsum[i] = gsum[i] / fmaxf(gcnt[i / FEAT], 1.0f);   // sums -> means, in place
}

__global__ void gn_center_kernel(const float* __restrict__ x, const int* __restrict__ batch,
                                 const float* __restrict__ gmean, const float* __restrict__ mscale,
                                 float* __restrict__ cent, float* __restrict__ gvar, int n_nodes) {
    long long gid = (long long)blockIdx.x * blockDim.x + threadIdx.x;
    int node = (int)(gid >> 4);
    int c4   = ((int)gid & 15) * 4;
    if (node >= n_nodes) return;
    int g = batch[node];
    const float* xr = x + (long long)node * FEAT + c4;
    const float* mr = gmean + (long long)g * FEAT + c4;
    float* cr = cent + (long long)node * FEAT + c4;
    float* vr = gvar + (long long)g * FEAT + c4;
#pragma unroll
    for (int j = 0; j < 4; ++j) {
        float c = xr[j] - mr[j] * mscale[c4 + j];
        cr[j] = c;
        atomicAdd(&vr[j], c * c);
    }
}

__global__ void gn_varfin_kernel(float* __restrict__ gvar, const float* __restrict__ gcnt, int total) {
    int i = blockIdx.x * blockDim.x + threadIdx.x;
    if (i >= total) return;
    gvar[i] = gvar[i] / fmaxf(gcnt[i / FEAT], 1.0f);
}

__global__ void gn_norm_kernel(const float* __restrict__ cent, const int* __restrict__ batch,
                               const float* __restrict__ gvar, const float* __restrict__ gw,
                               const float* __restrict__ gb, float* __restrict__ xout, int n_nodes) {
    long long gid = (long long)blockIdx.x * blockDim.x + threadIdx.x;
    int node = (int)(gid >> 4);
    int c4   = ((int)gid & 15) * 4;
    if (node >= n_nodes) return;
    int g = batch[node];
    const float* cr = cent + (long long)node * FEAT + c4;
    const float* vr = gvar + (long long)g * FEAT + c4;
    float* xo = xout + (long long)node * FEAT + c4;
#pragma unroll
    for (int j = 0; j < 4; ++j)
        xo[j] = gw[c4 + j] * cr[j] * rsqrtf(vr[j] + GN_EPS) + gb[c4 + j];
}

// ---------------- SoftmaxAggregation ----------------
__device__ __forceinline__ unsigned fenc(float f) {
    unsigned u = __float_as_uint(f);
    return (u & 0x80000000u) ? ~u : (u | 0x80000000u);
}
__device__ __forceinline__ float fdec(unsigned e) {
    unsigned u = (e & 0x80000000u) ? (e & 0x7FFFFFFFu) : ~e;
    return __uint_as_float(u);
}

__global__ void smax_init_kernel(unsigned* __restrict__ gmax, int total) {
    int i = blockIdx.x * blockDim.x + threadIdx.x;
    if (i >= total) return;
    gmax[i] = fenc(-FLT_MAX);
}

__global__ void smax_max_kernel(const float* __restrict__ x, const int* __restrict__ batch,
                                const float* __restrict__ t, unsigned* __restrict__ gmax, int n_nodes) {
    long long gid = (long long)blockIdx.x * blockDim.x + threadIdx.x;
    int node = (int)(gid >> 4);
    int c4   = ((int)gid & 15) * 4;
    if (node >= n_nodes) return;
    int g = batch[node];
    float tv = t[0];
    const float* xr = x + (long long)node * FEAT + c4;
    unsigned* mr = gmax + (long long)g * FEAT + c4;
#pragma unroll
    for (int j = 0; j < 4; ++j)
        atomicMax(&mr[j], fenc(tv * xr[j]));
}

__global__ void smax_sum_kernel(const float* __restrict__ x, const int* __restrict__ batch,
                                const float* __restrict__ t, const unsigned* __restrict__ gmax,
                                float* __restrict__ gden, float* __restrict__ gnum, int n_nodes) {
    long long gid = (long long)blockIdx.x * blockDim.x + threadIdx.x;
    int node = (int)(gid >> 4);
    int c4   = ((int)gid & 15) * 4;
    if (node >= n_nodes) return;
    int g = batch[node];
    float tv = t[0];
    const float* xr = x + (long long)node * FEAT + c4;
    const unsigned* mr = gmax + (long long)g * FEAT + c4;
    float* dr = gden + (long long)g * FEAT + c4;
    float* nr = gnum + (long long)g * FEAT + c4;
#pragma unroll
    for (int j = 0; j < 4; ++j) {
        float xv = xr[j];
        float w  = expf(tv * xv - fdec(mr[j]));
        atomicAdd(&dr[j], w);
        atomicAdd(&nr[j], w * xv);
    }
}

__global__ void smax_final_kernel(const float* __restrict__ gnum, const float* __restrict__ gden,
                                  float* __restrict__ out, int total) {
    int i = blockIdx.x * blockDim.x + threadIdx.x;
    if (i >= total) return;
    out[i] = gnum[i] / gden[i];
}

// ---------------- launch ----------------
extern "C" void kernel_launch(void* const* d_in, const int* in_sizes, int n_in,
                              void* d_out, int out_size, void* d_ws, size_t ws_size,
                              hipStream_t stream) {
    const float* x     = (const float*)d_in[0];
    const int*   ei    = (const int*)d_in[1];
    const int*   batch = (const int*)d_in[2];
    const float* Wl    = (const float*)d_in[3];
    const float* bl    = (const float*)d_in[4];
    const float* Wr    = (const float*)d_in[5];
    const float* gw    = (const float*)d_in[6];
    const float* gb    = (const float*)d_in[7];
    const float* gms   = (const float*)d_in[8];
    const float* t     = (const float*)d_in[9];

    const int n_nodes  = in_sizes[0] / FEAT;
    const int n_edges  = in_sizes[1] / 2;
    const int n_graphs = out_size / FEAT;
    const int n_layers = in_sizes[3] / (FEAT * FEAT);
    const int* src = ei;
    const int* dst = ei + n_edges;

    // workspace carve-out
    char* ws = (char*)d_ws;
    size_t off = 0;
    auto take = [&](size_t bytes) -> void* {
        void* p = ws + off;
        off += (bytes + 255) & ~(size_t)255;
        return p;
    };
    const size_t nodebytes = (size_t)n_nodes * FEAT * sizeof(float);
    const size_t gcbytes   = (size_t)n_graphs * FEAT * sizeof(float);
    float*    h    = (float*)take(nodebytes);       // node features (in-place across layers)
    float*    agg  = (float*)take(nodebytes);       // edge-aggregate, later reused for centered values
    float*    dinv = (float*)take((size_t)n_nodes * sizeof(float));
    float*    gsum = (float*)take(gcbytes);         // sums -> means
    float*    gcnt = (float*)take((size_t)n_graphs * sizeof(float));
    float*    gvar = (float*)take(gcbytes);
    unsigned* gmax = (unsigned*)take(gcbytes);
    float*    gden = (float*)take(gcbytes);
    float*    gnum = (float*)take(gcbytes);

    const int NT = 256;
    const int per_node16 = (int)(((long long)n_nodes * 16 + NT - 1) / NT);
    const int per_gc     = (n_graphs * FEAT + NT - 1) / NT;

    // inverse in-degree (shared by all layers)
    hipMemsetAsync(dinv, 0, (size_t)n_nodes * sizeof(float), stream);
    deg_kernel<<<(n_edges + NT - 1) / NT, NT, 0, stream>>>(dst, dinv, n_edges);
    invdeg_kernel<<<(n_nodes + NT - 1) / NT, NT, 0, stream>>>(dinv, n_nodes);

    // SAGE layers; layer 0 reads x, then in-place on h
    const float* cur = x;
    const int tiles  = (n_nodes + 15) / 16;
    const int tiles_per_block = 8 * TILES_PER_WAVE;
    const int gblocks = (tiles + tiles_per_block - 1) / tiles_per_block;
    for (int L = 0; L < n_layers; ++L) {
        hipMemsetAsync(agg, 0, nodebytes, stream);
        scatter_kernel<<<(int)(((long long)n_edges * 16 + NT - 1) / NT), NT, 0, stream>>>(
            cur, src, dst, agg, n_edges);
        sage_gemm_kernel<<<gblocks, NT, 0, stream>>>(
            cur, agg, dinv,
            Wl + (size_t)L * FEAT * FEAT, bl + (size_t)L * FEAT, Wr + (size_t)L * FEAT * FEAT,
            h, n_nodes);
        cur = h;
    }

    // GraphNorm
    hipMemsetAsync(gsum, 0, gcbytes, stream);
    hipMemsetAsync(gcnt, 0, (size_t)n_graphs * sizeof(float), stream);
    hipMemsetAsync(gvar, 0, gcbytes, stream);
    gn_stats_kernel<<<per_node16, NT, 0, stream>>>(h, batch, gsum, gcnt, n_nodes);
    gn_mean_kernel<<<per_gc, NT, 0, stream>>>(gsum, gcnt, n_graphs * FEAT);
    gn_center_kernel<<<per_node16, NT, 0, stream>>>(h, batch, gsum, gms, agg /*centered*/, gvar, n_nodes);
    gn_varfin_kernel<<<per_gc, NT, 0, stream>>>(gvar, gcnt, n_graphs * FEAT);
    gn_norm_kernel<<<per_node16, NT, 0, stream>>>(agg, batch, gvar, gw, gb, h, n_nodes);

    // SoftmaxAggregation
    hipMemsetAsync(gden, 0, gcbytes, stream);
    hipMemsetAsync(gnum, 0, gcbytes, stream);
    smax_init_kernel<<<per_gc, NT, 0, stream>>>(gmax, n_graphs * FEAT);
    smax_max_kernel<<<per_node16, NT, 0, stream>>>(h, batch, t, gmax, n_nodes);
    smax_sum_kernel<<<per_node16, NT, 0, stream>>>(h, batch, t, gmax, gden, gnum, n_nodes);
    smax_final_kernel<<<per_gc, NT, 0, stream>>>(gnum, gden, (float*)d_out, n_graphs * FEAT);
}
